// W4A16LlamaAttention_83416854823016
// MI455X (gfx1250) — matrix operand order repaired and checked by reading, not compiled, not run on hardware
//
#include <hip/hip_runtime.h>
#include <math.h>

typedef __bf16 bf16;
typedef __attribute__((ext_vector_type(16))) __bf16 bf16x16;
typedef __attribute__((ext_vector_type(8)))  __bf16 bf16x8;
typedef __attribute__((ext_vector_type(8)))  float  f32x8;

constexpr int kB   = 2;
constexpr int kS   = 2048;
constexpr int kHID = 4096;
constexpr int kNH  = 32;
constexpr int kNKV = 8;
constexpr int kHD  = 128;
constexpr int kNQKV = kNH*kHD + 2*kNKV*kHD;   // 6144
constexpr int kM    = kB*kS;                  // 4096 tokens

// ---------------- workspace layout (bytes) ----------------
constexpr size_t OFF_XP   = 0;                                     // X A-frags     bf16 4096x4096
constexpr size_t OFF_WQKV = OFF_XP   + (size_t)kM*kHID*2;          // Wqkv B-frags  bf16 4096x6144
constexpr size_t OFF_WO   = OFF_WQKV + (size_t)kHID*kNQKV*2;       // Wo   B-frags  bf16 4096x4096
constexpr size_t OFF_QKV  = OFF_WO   + (size_t)kHID*kHID*2;        // qkv row-major bf16 4096x6144
constexpr size_t OFF_QP   = OFF_QKV  + (size_t)kM*kNQKV*2;         // Q A-frags
constexpr size_t OFF_KP   = OFF_QP   + (size_t)kB*kNH*kS*kHD*2;    // K^T B-frags
constexpr size_t OFF_VP   = OFF_KP   + (size_t)kB*kNKV*kS*kHD*2;   // V B-frags
constexpr size_t OFF_CP   = OFF_VP   + (size_t)kB*kNKV*kS*kHD*2;   // ctx A-frags

// ---------------- WMMA wrapper ----------------
__device__ inline f32x8 wmma_bf16(bf16x16 a, bf16x16 b, f32x8 c) {
  return __builtin_amdgcn_wmma_f32_16x16x32_bf16(false, a, false, b, (short)0, c,
                                                 false, false);
}

// A-fragment element mapping (16x32 bf16 tile, ISA 7.12.2):
//   lane l: row = l&15, half = l>>4
//   elems 0..7  -> k = half*8 + e          (contiguous 16B)
//   elems 8..15 -> k = 16 + half*8 + (e-8) (contiguous 16B)
// B-fragment element mapping (32x16 bf16 tile):
//   lane l: col = l&15, half = l>>4, elem e -> k = half*8 + (e&7) + 16*(e>>3) (same as A, two contiguous 16B)

// ================= pack hidden_states (f32 -> bf16 A-frags) =================
__global__ void pack_x_kernel(const float* __restrict__ X, bf16* __restrict__ Xp) {
  int frag = blockIdx.x * (blockDim.x >> 5) + (threadIdx.x >> 5);
  int lane = threadIdx.x & 31;
  const int ktTot = kHID / 32;                 // 128
  int kt = frag % ktTot;
  int mt = frag / ktTot;
  if (mt >= kM / 16) return;
  int row = mt * 16 + (lane & 15);
  int hl  = lane >> 4;
  alignas(32) bf16 tmp[16];
  #pragma unroll
  for (int e = 0; e < 16; ++e) {
    int k = kt * 32 + hl * 8 + (e & 7) + ((e >> 3) * 16);
    tmp[e] = (bf16)X[(size_t)row * kHID + k];
  }
  ((bf16x16*)(Xp + (size_t)frag * 512))[lane] = *(const bf16x16*)tmp;
}

// ================= dequant int4 weights -> bf16 B-frags =================
__global__ void dequant_pack_kernel(const int* __restrict__ qw,
                                    const float* __restrict__ scale,
                                    bf16* __restrict__ Wp,
                                    int Kdim, int NtTot, int ntOff, int NtLoc) {
  int fragLocal = blockIdx.x * (blockDim.x >> 5) + (threadIdx.x >> 5);
  int lane = threadIdx.x & 31;
  int ktTot = Kdim / 32;
  int nt = fragLocal % NtLoc;
  int kt = fragLocal / NtLoc;
  if (kt >= ktTot) return;
  int hl = lane >> 4;
  int n  = nt * 16 + (lane & 15);              // local output row of W[out,in]
  const int*   qrow = qw    + (size_t)n * Kdim;
  const float* srow = scale + (size_t)n * (Kdim / 128);
  alignas(32) bf16 tmp[16];
  #pragma unroll
  for (int e = 0; e < 16; ++e) {
    int k = kt * 32 + hl * 8 + (e & 7) + ((e >> 3) * 16);
    tmp[e] = (bf16)((float)(qrow[k] - 8) * srow[k >> 7]);
  }
  size_t frag = (size_t)kt * NtTot + (ntOff + nt);
  ((bf16x16*)(Wp + frag * 512))[lane] = *(const bf16x16*)tmp;
}

// ================= fragment GEMM: C[M,N] = A x B =================
// 8 waves/block as 2(M) x 4(N); wave tile 32x64 (2x4 16-tiles) -> block tile 64x256.
// 8 WMMA per 6 fragment loads per K-step.
template <bool OUTF32>
__global__ void __launch_bounds__(256, 1)
gemm_kernel(const bf16* __restrict__ Ap, const bf16* __restrict__ Bp,
            void* __restrict__ Cout, int Ndim, int Kdim) {
  int wave = threadIdx.x >> 5;
  int lane = threadIdx.x & 31;
  int wm = wave >> 2;                          // 0..1
  int wn = wave & 3;                           // 0..3
  int m0t = blockIdx.y * 4  + wm * 2;          // 16-row tile index
  int n0t = blockIdx.x * 16 + wn * 4;          // 16-col tile index
  int ktTot = Kdim / 32;
  int ntTot = Ndim / 16;
  const bf16x16* Ab = (const bf16x16*)Ap;
  const bf16x16* Bb = (const bf16x16*)Bp;
  f32x8 acc[2][4] = {};
  for (int kt = 0; kt < ktTot; ++kt) {
    bf16x16 a0 = Ab[((size_t)m0t       * ktTot + kt) * 32 + lane];
    bf16x16 a1 = Ab[((size_t)(m0t + 1) * ktTot + kt) * 32 + lane];
    bf16x16 b0 = Bb[((size_t)kt * ntTot + n0t)     * 32 + lane];
    bf16x16 b1 = Bb[((size_t)kt * ntTot + n0t + 1) * 32 + lane];
    bf16x16 b2 = Bb[((size_t)kt * ntTot + n0t + 2) * 32 + lane];
    bf16x16 b3 = Bb[((size_t)kt * ntTot + n0t + 3) * 32 + lane];
    if (kt + 1 < ktTot) {
      __builtin_prefetch(&Ab[((size_t)m0t * ktTot + kt + 1) * 32 + lane], 0, 0);
      __builtin_prefetch(&Bb[((size_t)(kt + 1) * ntTot + n0t) * 32 + lane], 0, 0);
    }
    acc[0][0] = wmma_bf16(a0, b0, acc[0][0]);
    acc[0][1] = wmma_bf16(a0, b1, acc[0][1]);
    acc[0][2] = wmma_bf16(a0, b2, acc[0][2]);
    acc[0][3] = wmma_bf16(a0, b3, acc[0][3]);
    acc[1][0] = wmma_bf16(a1, b0, acc[1][0]);
    acc[1][1] = wmma_bf16(a1, b1, acc[1][1]);
    acc[1][2] = wmma_bf16(a1, b2, acc[1][2]);
    acc[1][3] = wmma_bf16(a1, b3, acc[1][3]);
  }
  int hl = lane >> 4, col = lane & 15;
  #pragma unroll
  for (int i = 0; i < 2; ++i)
    #pragma unroll
    for (int j = 0; j < 4; ++j)
      #pragma unroll
      for (int r = 0; r < 8; ++r) {
        int row = (m0t + i) * 16 + r + hl * 8;
        int c   = (n0t + j) * 16 + col;
        if (OUTF32) ((float*)Cout)[(size_t)row * Ndim + c] = acc[i][j][r];
        else        ((bf16*)Cout)[(size_t)row * Ndim + c]  = (bf16)acc[i][j][r];
      }
}

// ================= RoPE on q,k halves =================
__global__ void rope_kernel(bf16* __restrict__ qkv,
                            const float* __restrict__ cosb,
                            const float* __restrict__ sinb) {
  size_t idx = (size_t)blockIdx.x * blockDim.x + threadIdx.x;
  int d = idx & 63;
  size_t t = idx >> 6;
  int head  = (int)(t % (kNH + kNKV));         // 0..39  (32 q heads + 8 k heads)
  int token = (int)(t / (kNH + kNKV));
  if (token >= kM) return;
  int s  = token & (kS - 1);
  int bb = token >> 11;
  int col0 = head < kNH ? head * kHD : kNH * kHD + (head - kNH) * kHD;
  bf16* p = qkv + (size_t)token * kNQKV + col0;
  float x1 = (float)p[d], x2 = (float)p[d + 64];
  const float* cr = cosb + ((size_t)bb * kS + s) * kHD;
  const float* sr = sinb + ((size_t)bb * kS + s) * kHD;
  p[d]      = (bf16)(x1 * cr[d]      - x2 * sr[d]);
  p[d + 64] = (bf16)(x2 * cr[d + 64] + x1 * sr[d + 64]);
}

// ================= pack Q (A-frags), K^T and V (B-frags) =================
__global__ void pack_q_kernel(const bf16* __restrict__ qkv, bf16* __restrict__ Qp) {
  int frag = blockIdx.x * (blockDim.x >> 5) + (threadIdx.x >> 5);
  int lane = threadIdx.x & 31;
  int dt = frag & 3;
  int rest = frag >> 2;
  int qt = rest & (kS / 16 - 1);  rest >>= 7;
  int h  = rest % kNH;
  int b  = rest / kNH;
  if (b >= kB) return;
  int srow = qt * 16 + (lane & 15);
  int hl = lane >> 4;
  const bf16* base = qkv + ((size_t)(b * kS + srow)) * kNQKV + h * kHD;
  alignas(32) bf16 tmp[16];
  #pragma unroll
  for (int e = 0; e < 16; ++e) {
    int d = dt * 32 + hl * 8 + (e & 7) + ((e >> 3) * 16);
    tmp[e] = base[d];
  }
  ((bf16x16*)(Qp + (size_t)frag * 512))[lane] = *(const bf16x16*)tmp;
}

__global__ void pack_k_kernel(const bf16* __restrict__ qkv, bf16* __restrict__ Kp) {
  int frag = blockIdx.x * (blockDim.x >> 5) + (threadIdx.x >> 5);
  int lane = threadIdx.x & 31;
  int st = frag & (kS / 16 - 1);
  int rest = frag >> 7;
  int dt = rest & 3;  rest >>= 2;
  int h  = rest % kNKV;
  int b  = rest / kNKV;
  if (b >= kB) return;
  int key = st * 16 + (lane & 15);
  int hl = lane >> 4;
  const bf16* base = qkv + ((size_t)(b * kS + key)) * kNQKV + kNH * kHD + h * kHD;
  alignas(32) bf16 tmp[16];
  #pragma unroll
  for (int e = 0; e < 16; ++e) {
    int d = dt * 32 + hl * 8 + (e & 7) + ((e >> 3) * 16);  // B-frag: k-dim = head dim
    tmp[e] = base[d];
  }
  ((bf16x16*)(Kp + (size_t)frag * 512))[lane] = *(const bf16x16*)tmp;
}

__global__ void pack_v_kernel(const bf16* __restrict__ qkv, bf16* __restrict__ Vp) {
  int frag = blockIdx.x * (blockDim.x >> 5) + (threadIdx.x >> 5);
  int lane = threadIdx.x & 31;
  int dt = frag & 7;
  int rest = frag >> 3;
  int st = rest & (kS / 32 - 1);  rest >>= 6;
  int h  = rest % kNKV;
  int b  = rest / kNKV;
  if (b >= kB) return;
  int hl = lane >> 4;
  int dcol = dt * 16 + (lane & 15);
  alignas(32) bf16 tmp[16];
  #pragma unroll
  for (int e = 0; e < 16; ++e) {
    int key = st * 32 + hl * 8 + (e & 7) + ((e >> 3) * 16);  // B-frag: k-dim = key
    tmp[e] = qkv[((size_t)(b * kS + key)) * kNQKV + (kNH + kNKV) * kHD + h * kHD + dcol];
  }
  ((bf16x16*)(Vp + (size_t)frag * 512))[lane] = *(const bf16x16*)tmp;
}

// ================= flash attention: one wave per 16-query tile =================
__global__ void __launch_bounds__(256, 1)
attn_kernel(const bf16* __restrict__ Qp, const bf16* __restrict__ Kp,
            const bf16* __restrict__ Vp, bf16* __restrict__ Cp) {
  __shared__ bf16 lds[8 * 2048];                 // 4KB per wave
  int wave = threadIdx.x >> 5;
  int lane = threadIdx.x & 31;
  int bh = blockIdx.x;
  int h  = bh % kNH;
  int b  = bh / kNH;
  int kvh = h >> 2;                              // NH/NKV = 4
  int qt  = blockIdx.y * 8 + wave;               // 16-query tile index within sequence
  bf16* myLds = lds + wave * 2048;
  int hl = lane >> 4, col = lane & 15;

  const bf16x16* Qb = (const bf16x16*)Qp;
  const bf16x16* Kb = (const bf16x16*)Kp;
  const bf16x16* Vb = (const bf16x16*)Vp;

  bf16x16 qf[4];
  size_t qfrag0 = ((size_t)(b * kNH + h) * (kS / 16) + qt) * 4;
  #pragma unroll
  for (int dt = 0; dt < 4; ++dt) qf[dt] = Qb[(qfrag0 + dt) * 32 + lane];

  f32x8 ctx[8] = {};
  float mrow[8], lrow[8];
  #pragma unroll
  for (int r = 0; r < 8; ++r) { mrow[r] = -3.0e38f; lrow[r] = 0.f; }

  const float scale = 0.08838834764831845f;      // HD^-0.5
  size_t kbase = (size_t)(b * kNKV + kvh) * 4 * (kS / 16);
  size_t vbase = (size_t)(b * kNKV + kvh) * (kS / 32) * 8;

  int stEnd = (qt * 16 + 15) >> 5;               // inclusive: causal upper bound
  for (int st = 0; st <= stEnd; ++st) {
    // ---- S = Q K^T  (two 16x16 n-tiles over 32 keys) ----
    f32x8 s0 = {}, s1 = {};
    #pragma unroll
    for (int dt = 0; dt < 4; ++dt) {
      bf16x16 k0 = Kb[(kbase + (size_t)dt * (kS / 16) + st * 2)     * 32 + lane];
      bf16x16 k1 = Kb[(kbase + (size_t)dt * (kS / 16) + st * 2 + 1) * 32 + lane];
      s0 = wmma_bf16(qf[dt], k0, s0);
      s1 = wmma_bf16(qf[dt], k1, s1);
    }
    // ---- mask + online softmax; stream P rows straight into LDS ----
    int qrow_base = qt * 16 + hl * 8;
    #pragma unroll
    for (int r = 0; r < 8; ++r) {
      int qrow = qrow_base + r;
      int key0 = st * 32 + col;
      float v0 = s0[r] * scale + (key0      <= qrow ? 0.f : -1.0e9f);
      float v1 = s1[r] * scale + (key0 + 16 <= qrow ? 0.f : -1.0e9f);
      float mx = fmaxf(v0, v1);
      #pragma unroll
      for (int off = 1; off < 16; off <<= 1) mx = fmaxf(mx, __shfl_xor(mx, off, 32));
      float mnew = fmaxf(mrow[r], mx);
      float cf = __expf(mrow[r] - mnew);
      float p0 = __expf(v0 - mnew);
      float p1 = __expf(v1 - mnew);
      float ps = p0 + p1;
      #pragma unroll
      for (int off = 1; off < 16; off <<= 1) ps += __shfl_xor(ps, off, 32);
      lrow[r] = lrow[r] * cf + ps;
      mrow[r] = mnew;
      #pragma unroll
      for (int dc = 0; dc < 8; ++dc) ctx[dc][r] *= cf;
      int row = hl * 8 + r;                      // P tile row (C layout)
      myLds[row * 32 + col]      = (bf16)p0;
      myLds[row * 32 + col + 16] = (bf16)p1;
    }
    asm volatile("s_wait_dscnt 0x0" ::: "memory");
    // ---- P (C layout) -> A-frag layout ----
    bf16x8 lo = *(const bf16x8*)(myLds + col * 32 + hl * 8);
    bf16x8 hi = *(const bf16x8*)(myLds + col * 32 + hl * 8 + 16);
    bf16x16 pa = __builtin_shufflevector(lo, hi, 0, 1, 2, 3, 4, 5, 6, 7,
                                         8, 9, 10, 11, 12, 13, 14, 15);
    // ---- ctx += P V ----
    #pragma unroll
    for (int dc = 0; dc < 8; ++dc) {
      bf16x16 vf = Vb[(vbase + (size_t)st * 8 + dc) * 32 + lane];
      ctx[dc] = wmma_bf16(pa, vf, ctx[dc]);
    }
  }

  // ---- normalize, then emit ctx as A-frags for the O-projection GEMM ----
  #pragma unroll
  for (int dc = 0; dc < 8; ++dc)
    #pragma unroll
    for (int r = 0; r < 8; ++r) {
      float val = ctx[dc][r] / lrow[r];
      myLds[(hl * 8 + r) * 128 + dc * 16 + col] = (bf16)val;
    }
  asm volatile("s_wait_dscnt 0x0" ::: "memory");
  size_t mtile = (size_t)b * (kS / 16) + qt;     // token tile
  #pragma unroll
  for (int dt = 0; dt < 4; ++dt) {
    bf16x8 lo = *(const bf16x8*)(myLds + col * 128 + dt * 32 + hl * 8);
    bf16x8 hi = *(const bf16x8*)(myLds + col * 128 + dt * 32 + hl * 8 + 16);
    bf16x16 af = __builtin_shufflevector(lo, hi, 0, 1, 2, 3, 4, 5, 6, 7,
                                         8, 9, 10, 11, 12, 13, 14, 15);
    size_t frag = mtile * (kHID / 32) + (size_t)h * 4 + dt;
    ((bf16x16*)Cp)[frag * 32 + lane] = af;
  }
}

// ================= host launcher =================
extern "C" void kernel_launch(void* const* d_in, const int* in_sizes, int n_in,
                              void* d_out, int out_size, void* d_ws, size_t ws_size,
                              hipStream_t stream) {
  (void)in_sizes; (void)n_in; (void)out_size; (void)ws_size;
  const float* hs      = (const float*)d_in[0];
  const float* cosb    = (const float*)d_in[1];
  const float* sinb    = (const float*)d_in[2];
  // d_in[3] = attention_mask (causal mask computed inline)
  const int*   q_qw    = (const int*)d_in[4];
  const float* q_scale = (const float*)d_in[5];
  const int*   k_qw    = (const int*)d_in[6];
  const float* k_scale = (const float*)d_in[7];
  const int*   v_qw    = (const int*)d_in[8];
  const float* v_scale = (const float*)d_in[9];
  const int*   o_qw    = (const int*)d_in[10];
  const float* o_scale = (const float*)d_in[11];

  char* ws = (char*)d_ws;
  bf16* Xp    = (bf16*)(ws + OFF_XP);
  bf16* Wqkv  = (bf16*)(ws + OFF_WQKV);
  bf16* Wo    = (bf16*)(ws + OFF_WO);
  bf16* qkv   = (bf16*)(ws + OFF_QKV);
  bf16* Qp    = (bf16*)(ws + OFF_QP);
  bf16* Kp    = (bf16*)(ws + OFF_KP);
  bf16* Vp    = (bf16*)(ws + OFF_VP);
  bf16* Cp    = (bf16*)(ws + OFF_CP);
  float* out  = (float*)d_out;

  // weight dequant + fragment pack (frags/block = 8 with 256 threads)
  dequant_pack_kernel<<<(128 * 256) / 8, 256, 0, stream>>>(q_qw, q_scale, Wqkv, kHID, 384, 0,   256);
  dequant_pack_kernel<<<(128 * 64)  / 8, 256, 0, stream>>>(k_qw, k_scale, Wqkv, kHID, 384, 256, 64);
  dequant_pack_kernel<<<(128 * 64)  / 8, 256, 0, stream>>>(v_qw, v_scale, Wqkv, kHID, 384, 320, 64);
  dequant_pack_kernel<<<(128 * 256) / 8, 256, 0, stream>>>(o_qw, o_scale, Wo,   kHID, 256, 0,   256);

  // activation pack
  pack_x_kernel<<<(256 * 128) / 8, 256, 0, stream>>>(hs, Xp);

  // QKV projection: [4096 x 4096] x [4096 x 6144] -> bf16
  gemm_kernel<false><<<dim3(kNQKV / 256, kM / 64), 256, 0, stream>>>(Xp, Wqkv, qkv, kNQKV, kHID);

  // RoPE on q and k
  {
    size_t total = (size_t)kM * (kNH + kNKV) * 64;
    rope_kernel<<<(unsigned)(total / 256), 256, 0, stream>>>(qkv, cosb, sinb);
  }

  // per-head fragment packs
  pack_q_kernel<<<(kB * kNH * (kS / 16) * 4) / 8, 256, 0, stream>>>(qkv, Qp);
  pack_k_kernel<<<(kB * kNKV * 4 * (kS / 16)) / 8, 256, 0, stream>>>(qkv, Kp);
  pack_v_kernel<<<(kB * kNKV * (kS / 32) * 8) / 8, 256, 0, stream>>>(qkv, Vp);

  // attention (flash, causal) -> ctx A-frags
  attn_kernel<<<dim3(kB * kNH, kS / 16 / 8), 256, 0, stream>>>(Qp, Kp, Vp, Cp);

  // O projection: [4096 x 4096] x [4096 x 4096] -> f32 out
  gemm_kernel<true><<<dim3(kHID / 256, kM / 64), 256, 0, stream>>>(Cp, Wo, out, kHID, kHID);
}
